// DualSlidingWindowAttention_35570919145907
// MI455X (gfx1250) — matrix operand order repaired and checked by reading, not compiled, not run on hardware
//
#include <hip/hip_runtime.h>

#define T_LEN   1024
#define HIDDEN  2048
#define NBATCH  2
#define NHQ     32
#define NHKV    8
#define DHEAD   64
#define KVDIM   512
#define WIN     256
#define QT      16
#define SKV     288            // padded kv columns per half (window 256 + 32, 32-aligned)
#define CCOL    (2*SKV)        // 576 total kv columns (ssm half | attn half)
#define CDW     (CCOL/2)       // 288 dwords per P row

typedef __attribute__((ext_vector_type(16))) __bf16 v16bf;
typedef __attribute__((ext_vector_type(8)))  float  v8f;
typedef __attribute__((ext_vector_type(4)))  unsigned int v4u;
typedef __attribute__((ext_vector_type(8)))  unsigned int v8u;

#define AS3 __attribute__((address_space(3)))

union AF { v16bf v; unsigned int u[8]; };

__device__ __forceinline__ unsigned short f2bf(float x) {
  union { float f; unsigned int u; } c; c.f = x;
  unsigned int u = c.u;
  if ((u & 0x7fffffffu) > 0x7f800000u) return (unsigned short)((u >> 16) | 0x0040u); // NaN
  u += 0x7fffu + ((u >> 16) & 1u);   // round-to-nearest-even
  return (unsigned short)(u >> 16);
}

// LDS byte offset of a __shared__ object (generic -> LDS addrspace -> offset)
__device__ __forceinline__ unsigned lds_off(const void* p) {
  return (unsigned)(unsigned long long)(AS3 const void*)p;
}

// Issue a TDM 2D tile load: tile_dim0=32 elems (K), tile_dim1=64 rows, bf16 (2B),
// row stride = K elements. D# per CDNA5 ISA 8.3/8.4 (group0: count/lds/global/type,
// group1: data_size, dims, tile dims, stride). Tracked by TENSORcnt.
__device__ __forceinline__ void tdm_load_tile(unsigned long long gaddr, unsigned lds,
                                              unsigned Kelems) {
  v4u g0;
  g0[0] = 1u;                                                 // count=1, no gather
  g0[1] = lds;                                                // lds_addr
  g0[2] = (unsigned)(gaddr & 0xffffffffu);                    // global_addr[31:0]
  g0[3] = (unsigned)((gaddr >> 32) & 0x1ffffffu) | (2u << 30);// addr[56:32] | type=2
  const unsigned DIM1 = 1u << 20;                             // large: no OOB clip
  v8u g1;
  g1[0] = 1u << 16;                                           // wg_mask=0, data_size=2B
  g1[1] = (Kelems & 0xffffu) << 16;                           // tensor_dim0[15:0]
  g1[2] = ((Kelems >> 16) & 0xffffu) | ((DIM1 & 0xffffu) << 16); // dim0 hi | dim1 lo
  g1[3] = ((DIM1 >> 16) & 0xffffu) | (32u << 16);             // dim1 hi | tile_dim0=32
  g1[4] = 64u;                                                // tile_dim1=64, tile_dim2=0
  g1[5] = Kelems;                                             // tensor_dim0_stride[31:0]
  g1[6] = 0u;                                                 // stride0 hi | dim1_stride lo
  g1[7] = 0u;
  asm volatile("tensor_load_to_lds %0, %1" : : "s"(g0), "s"(g1) : "memory");
}

__global__ void cvt_bf16_kernel(const float* __restrict__ src,
                                unsigned short* __restrict__ dst, int n4) {
  int i = blockIdx.x * blockDim.x + threadIdx.x;
  if (i < n4) {
    float4 f = ((const float4*)src)[i];
    uint2 o;
    o.x = (unsigned)f2bf(f.x) | ((unsigned)f2bf(f.y) << 16);
    o.y = (unsigned)f2bf(f.z) | ((unsigned)f2bf(f.w) << 16);
    ((uint2*)dst)[i] = o;
  }
}

// Wt[n*K + k] = bf16(W[k*N + n])  (weights pre-transposed to N-major for B-fragments)
__global__ void cvtT_bf16_kernel(const float* __restrict__ W,
                                 unsigned short* __restrict__ Wt, int K, int N) {
  int i = blockIdx.x * blockDim.x + threadIdx.x;   // over N*(K/4)
  int Kq = K >> 2;
  if (i < N * Kq) {
    int n = i / Kq, k4 = (i - n * Kq) << 2;
    uint2 o;
    o.x = (unsigned)f2bf(W[(size_t)(k4 + 0) * N + n]) |
          ((unsigned)f2bf(W[(size_t)(k4 + 1) * N + n]) << 16);
    o.y = (unsigned)f2bf(W[(size_t)(k4 + 2) * N + n]) |
          ((unsigned)f2bf(W[(size_t)(k4 + 3) * N + n]) << 16);
    *(uint2*)(Wt + (size_t)n * K + k4) = o;
  }
}

// C[M,N] = A[M,K] (bf16 row-major) @ Bt[N,K]^T (bf16 N-major).
// Tiles staged into LDS by the Tensor Data Mover (wave 0 issues, TENSORcnt waits).
// mode 0: bf16 row-major out; mode 1: bf16 V-transposed [b,hk,d,t]; mode 2: f32 row-major.
__global__ __launch_bounds__(128) void gemm_bf16_wmma(
    const unsigned short* __restrict__ A, const unsigned short* __restrict__ Bt,
    void* __restrict__ C, int M, int N, int K, int mode)
{
  __shared__ __align__(16) unsigned int lA[64 * 16];   // 64 rows x 32 bf16
  __shared__ __align__(16) unsigned int lB[64 * 16];
  const int m0 = blockIdx.x * 64;
  const int n0 = blockIdx.y * 64;
  const int tid = threadIdx.x;
  const int lane = tid & 31;
  const int wave = tid >> 5;
  const int lh   = lane >> 4;     // lane half (K-half selector)
  const int r    = lane & 15;

  const unsigned ldsA = lds_off(lA);
  const unsigned ldsB = lds_off(lB);
  const unsigned long long gA = (unsigned long long)(const void*)A;
  const unsigned long long gB = (unsigned long long)(const void*)Bt;

  v8f zero = {0.f,0.f,0.f,0.f,0.f,0.f,0.f,0.f};
  v8f acc[4] = {zero, zero, zero, zero};

  for (int k0 = 0; k0 < K; k0 += 32) {
    __syncthreads();                      // previous iteration's fragment reads done
    if (wave == 0) {
      tdm_load_tile(gA + ((size_t)m0 * (size_t)K + k0) * 2ull, ldsA, (unsigned)K);
      tdm_load_tile(gB + ((size_t)n0 * (size_t)K + k0) * 2ull, ldsB, (unsigned)K);
      __builtin_amdgcn_s_wait_tensorcnt(0);
    }
    __syncthreads();                      // DMA'd tiles visible to all waves
    AF bf;
    #pragma unroll
    for (int j = 0; j < 8; j++) bf.u[j] = lB[(wave * 16 + r) * 16 + lh * 8 + j];
    #pragma unroll
    for (int rs = 0; rs < 4; rs++) {
      AF af;
      #pragma unroll
      for (int j = 0; j < 8; j++)
        af.u[j] = lA[(rs * 16 + r) * 16 + lh * 8 + j];
      acc[rs] = __builtin_amdgcn_wmma_f32_16x16x32_bf16(
          false, af.v, false, bf.v, (short)0, acc[rs], false, false);
    }
  }

  #pragma unroll
  for (int rs = 0; rs < 4; rs++) {
    #pragma unroll
    for (int rr = 0; rr < 8; rr++) {
      int m = m0 + rs * 16 + rr + lh * 8;
      int n = n0 + wave * 16 + r;
      float val = acc[rs][rr];
      if (mode == 2) {
        ((float*)C)[(size_t)m * N + n] = val;
      } else if (mode == 1) {
        int b = m / T_LEN, t = m - b * T_LEN;
        int hk = n / DHEAD, d = n - hk * DHEAD;
        ((unsigned short*)C)[(((size_t)(b * NHKV + hk)) * DHEAD + d) * T_LEN + t] = f2bf(val);
      } else {
        ((unsigned short*)C)[(size_t)m * N + n] = f2bf(val);
      }
    }
  }
}

// One block = (b, query head h, 16-row query tile). 4 waves.
__global__ __launch_bounds__(128) void attn_wmma_kernel(
    const unsigned short* __restrict__ qB,   // [B*T][2048] bf16
    const unsigned short* __restrict__ kB,   // [B*T][512]  bf16
    const unsigned short* __restrict__ skB,  // [B*T][512]  bf16
    const unsigned short* __restrict__ vT,   // [B*HK*64][1024] bf16 (d-major)
    const unsigned short* __restrict__ svT,
    unsigned short* __restrict__ attnB)      // [B*T][2048] bf16
{
  __shared__ float        sS[QT * CCOL];     // raw scores (f32)
  __shared__ unsigned int sP[QT * CDW];      // probabilities (bf16 packed)

  const int blk = blockIdx.x;
  const int qt  = blk & 63;                  // T/QT = 64 tiles
  const int h   = (blk >> 6) & 31;
  const int b   = blk >> 11;
  const int hk  = h >> 2;                    // GQA: G = 4
  const int qt0 = qt * QT;
  const int kv0 = qt0 - WIN;                 // 16-aligned, may be negative

  const int tid  = threadIdx.x;
  const int lane = tid & 31;
  const int wave = tid >> 5;
  const int lh   = lane >> 4;
  const int r    = lane & 15;

  const unsigned int* qdw  = (const unsigned int*)qB;
  const unsigned int* kdw  = (const unsigned int*)kB;
  const unsigned int* skdw = (const unsigned int*)skB;

  // Q A-fragments for this tile's 16 rows: d 0..31 and 32..63 (b128 loads)
  AF qf[2];
  {
    size_t base = (size_t)(b * T_LEN + qt0 + r) * (HIDDEN / 2) + h * (DHEAD / 2) + lh * 8;
    #pragma unroll
    for (int kk = 0; kk < 2; kk++) {
      *(uint4*)&qf[kk].u[0] = *(const uint4*)(qdw + base + kk * 16);
      *(uint4*)&qf[kk].u[4] = *(const uint4*)(qdw + base + kk * 16 + 4);
    }
  }

  const float slope = exp2f(-0.25f * (float)h);   // 2^(-8h/32)
  const float scale = 0.125f;                     // 1/sqrt(64)
  v8f zero = {0.f,0.f,0.f,0.f,0.f,0.f,0.f,0.f};

  // ---- scores: 36 column tiles (18 per half), wave w takes cc = w + 4i ----
  for (int i = 0; i < 9; i++) {
    int cc   = wave + 4 * i;
    int hsel = (cc >= 18) ? 1 : 0;               // 0 = ssm half, 1 = attn half
    int ct   = cc - hsel * 18;
    const unsigned int* ksrc = hsel ? kdw : skdw;
    int s_row = kv0 + ct * 16 + r;
    int s_cl  = s_row < 0 ? 0 : (s_row > T_LEN - 1 ? T_LEN - 1 : s_row);
    size_t kbase = (size_t)(b * T_LEN + s_cl) * (KVDIM / 2) + hk * (DHEAD / 2) + lh * 8;
    v8f acc = zero;
    #pragma unroll
    for (int kk = 0; kk < 2; kk++) {
      AF bf;
      *(uint4*)&bf.u[0] = *(const uint4*)(ksrc + kbase + kk * 16);
      *(uint4*)&bf.u[4] = *(const uint4*)(ksrc + kbase + kk * 16 + 4);
      acc = __builtin_amdgcn_wmma_f32_16x16x32_bf16(
          false, qf[kk].v, false, bf.v, (short)0, acc, false, false);
    }
    int s_c = kv0 + ct * 16 + r;                 // C-frag column N == lane&15 == r
    #pragma unroll
    for (int rr = 0; rr < 8; rr++) {
      int tq = qt0 + rr + lh * 8;
      float rel = (float)(s_c - tq);
      bool valid = (s_c >= 0) && (rel <= 0.f) && (rel > -(float)WIN);
      float sc = valid ? acc[rr] * scale + slope * rel : -1e30f;
      sS[(rr + lh * 8) * CCOL + hsel * SKV + ct * 16 + r] = sc;
    }
  }
  __syncthreads();

  // ---- softmax over the 576 bounded columns (one thread per query row) ----
  if (tid < QT) {
    float mx = -3.0e38f;
    for (int j = 0; j < CCOL; j++) mx = fmaxf(mx, sS[tid * CCOL + j]);
    float sum = 0.f;
    for (int j = 0; j < CCOL; j++) {
      float e = __expf(sS[tid * CCOL + j] - mx);
      sS[tid * CCOL + j] = e;
      sum += e;
    }
    float inv = 1.f / sum;
    for (int j = 0; j < CDW; j++) {
      unsigned int lo = f2bf(sS[tid * CCOL + 2 * j] * inv);
      unsigned int hi = f2bf(sS[tid * CCOL + 2 * j + 1] * inv);
      sP[tid * CDW + j] = lo | (hi << 16);
    }
  }
  __syncthreads();

  // ---- O = P @ V : each wave owns one 16-wide d slice; K = 576 -> 18 WMMAs ----
  v8f oacc = zero;
  const int drow = (b * NHKV + hk) * DHEAD + wave * 16 + r;  // V^T row (n = d)
  for (int cc = 0; cc < 18; cc++) {
    int hsel = (cc >= 9) ? 1 : 0;
    int cb   = (cc - hsel * 9) * 32;             // chunk base within half (32-aligned)
    const unsigned int* vsrc = (const unsigned int*)(hsel ? vT : svT);
    AF pf, bf;
    #pragma unroll
    for (int j = 0; j < 8; j++)
      pf.u[j] = sP[r * CDW + hsel * (SKV / 2) + (cb >> 1) + lh * 8 + j];
    int sbase = kv0 + cb;                        // chunk-uniform in-range test
    if (sbase >= 0 && sbase + 31 < T_LEN) {
      const unsigned int* vp = vsrc + (size_t)drow * (T_LEN / 2) + (sbase >> 1) + lh * 8;
      *(uint4*)&bf.u[0] = *(const uint4*)(vp);
      *(uint4*)&bf.u[4] = *(const uint4*)(vp + 4);
    } else {
      #pragma unroll
      for (int j = 0; j < 8; j++) {
        int sp  = sbase + lh * 16 + 2 * j;       // even (kv0 16-aligned)
        int sp2 = sp < 0 ? 0 : (sp > T_LEN - 2 ? T_LEN - 2 : sp);  // masked by P==0
        bf.u[j] = vsrc[(size_t)drow * (T_LEN / 2) + (sp2 >> 1)];
      }
    }
    oacc = __builtin_amdgcn_wmma_f32_16x16x32_bf16(
        false, pf.v, false, bf.v, (short)0, oacc, false, false);
  }
  #pragma unroll
  for (int rr = 0; rr < 8; rr++) {
    int tq = qt0 + rr + lh * 8;
    attnB[(size_t)(b * T_LEN + tq) * HIDDEN + h * DHEAD + wave * 16 + r] = f2bf(oacc[rr]);
  }
}

extern "C" void kernel_launch(void* const* d_in, const int* in_sizes, int n_in,
                              void* d_out, int out_size, void* d_ws, size_t ws_size,
                              hipStream_t stream) {
  (void)in_sizes; (void)n_in; (void)out_size; (void)ws_size;
  const float* hs  = (const float*)d_in[0];
  const float* ssm = (const float*)d_in[1];
  const float* Wq  = (const float*)d_in[2];
  const float* Wk  = (const float*)d_in[3];
  const float* Wv  = (const float*)d_in[4];
  const float* Wsk = (const float*)d_in[5];
  const float* Wsv = (const float*)d_in[6];
  const float* Wo  = (const float*)d_in[7];
  float* out = (float*)d_out;

  char* base = (char*)d_ws;
  size_t off = 0;
  auto nxt = [&](size_t bytes) {
    char* p = base + off;
    off += (bytes + 255) & ~(size_t)255;
    return p;
  };

  const size_t M = (size_t)NBATCH * T_LEN;   // 2048
  unsigned short* hsb  = (unsigned short*)nxt(M * HIDDEN * 2);
  unsigned short* ssb  = (unsigned short*)nxt(M * HIDDEN * 2);
  unsigned short* Wqt  = (unsigned short*)nxt((size_t)HIDDEN * HIDDEN * 2);
  unsigned short* Wot  = (unsigned short*)nxt((size_t)HIDDEN * HIDDEN * 2);
  unsigned short* Wkt  = (unsigned short*)nxt((size_t)KVDIM * HIDDEN * 2);
  unsigned short* Wvt  = (unsigned short*)nxt((size_t)KVDIM * HIDDEN * 2);
  unsigned short* Wskt = (unsigned short*)nxt((size_t)KVDIM * HIDDEN * 2);
  unsigned short* Wsvt = (unsigned short*)nxt((size_t)KVDIM * HIDDEN * 2);
  unsigned short* qb   = (unsigned short*)nxt(M * HIDDEN * 2);
  unsigned short* kb   = (unsigned short*)nxt(M * KVDIM * 2);
  unsigned short* skb  = (unsigned short*)nxt(M * KVDIM * 2);
  unsigned short* vt   = (unsigned short*)nxt(M * KVDIM * 2);
  unsigned short* svt  = (unsigned short*)nxt(M * KVDIM * 2);
  unsigned short* ab   = (unsigned short*)nxt(M * HIDDEN * 2);

  int nfull4 = (int)(M * HIDDEN / 4);
  cvt_bf16_kernel<<<(nfull4 + 255) / 256, 256, 0, stream>>>(hs, hsb, nfull4);
  cvt_bf16_kernel<<<(nfull4 + 255) / 256, 256, 0, stream>>>(ssm, ssb, nfull4);
  int nqq = HIDDEN * HIDDEN / 4;
  int nkv = HIDDEN * KVDIM / 4;
  cvtT_bf16_kernel<<<(nqq + 255) / 256, 256, 0, stream>>>(Wq,  Wqt,  HIDDEN, HIDDEN);
  cvtT_bf16_kernel<<<(nqq + 255) / 256, 256, 0, stream>>>(Wo,  Wot,  HIDDEN, HIDDEN);
  cvtT_bf16_kernel<<<(nkv + 255) / 256, 256, 0, stream>>>(Wk,  Wkt,  HIDDEN, KVDIM);
  cvtT_bf16_kernel<<<(nkv + 255) / 256, 256, 0, stream>>>(Wv,  Wvt,  HIDDEN, KVDIM);
  cvtT_bf16_kernel<<<(nkv + 255) / 256, 256, 0, stream>>>(Wsk, Wskt, HIDDEN, KVDIM);
  cvtT_bf16_kernel<<<(nkv + 255) / 256, 256, 0, stream>>>(Wsv, Wsvt, HIDDEN, KVDIM);

  dim3 blk(128);
  gemm_bf16_wmma<<<dim3(32, 32), blk, 0, stream>>>(hsb, Wqt,  qb,  2048, 2048, 2048, 0);
  gemm_bf16_wmma<<<dim3(32, 8),  blk, 0, stream>>>(hsb, Wkt,  kb,  2048, 512,  2048, 0);
  gemm_bf16_wmma<<<dim3(32, 8),  blk, 0, stream>>>(ssb, Wskt, skb, 2048, 512,  2048, 0);
  gemm_bf16_wmma<<<dim3(32, 8),  blk, 0, stream>>>(hsb, Wvt,  vt,  2048, 512,  2048, 1);
  gemm_bf16_wmma<<<dim3(32, 8),  blk, 0, stream>>>(ssb, Wsvt, svt, 2048, 512,  2048, 1);

  attn_wmma_kernel<<<NBATCH * NHQ * (T_LEN / QT), 128, 0, stream>>>(qb, kb, skb, vt, svt, ab);

  gemm_bf16_wmma<<<dim3(32, 32), blk, 0, stream>>>(ab, Wot, out, 2048, 2048, 2048, 2);
}
